// GeomEncoder_5420248728166
// MI455X (gfx1250) — matrix operand order repaired and checked by reading, not compiled, not run on hardware
//
#include <hip/hip_runtime.h>

typedef _Float16 f16;
typedef __attribute__((ext_vector_type(16))) _Float16 v16h;
typedef __attribute__((ext_vector_type(8)))  _Float16 h8;
typedef __attribute__((ext_vector_type(8)))  float    v8f;

#define NN   100   // real nodes
#define NP   112   // padded rows (7*16)
#define NP2  128   // padded attention K (4*32)
#define DD   256   // feature dim
#define HS   264   // hs row stride (halves)
#define GST  136   // gsT row stride (halves), rows are features o, cols are nodes j
#define AS   136   // attn row stride (halves)
#define MT   7     // row tiles (16 rows each)
#define CTN  16    // col tiles (16 cols each)
#define KT8  8     // K steps of 32 over D=256
#define AKT  4     // K steps of 32 over NP2=128

struct GatParams {
  const float* x;
  const float* rw;   // remap_w [256][6]
  const float* rb;   // remap_b [256]
  const f16*   wt;   // fragment-packed f16 weights, 4 * 65536 halves
  const float* as0; const float* ad0; const float* bb0;
  const float* as1; const float* ad1; const float* bb1;
  const float* as2; const float* ad2; const float* bb2;
  const float* as3; const float* ad3; const float* bb3;
  float* out;
};

// Pack w_l [o=256][k=256] f32 into B-fragment order per (kt,ct) tile:
// lane l holds 16 contiguous halves: w[ct*16 + (l&15)][kt*32 + (l>>4)*16 + i]
__global__ __launch_bounds__(256) void gat_prepack(const float* __restrict__ w0,
                                                   const float* __restrict__ w1,
                                                   const float* __restrict__ w2,
                                                   const float* __restrict__ w3,
                                                   f16* __restrict__ wt) {
  int gid   = blockIdx.x * 256 + threadIdx.x;   // 0 .. 262143
  int layer = gid >> 16;
  int rem   = gid & 65535;
  int bi    = rem >> 9;          // tile id: kt*16 + ct
  int lane  = (rem >> 4) & 31;
  int i     = rem & 15;
  int kt = bi >> 4, ct = bi & 15;
  int col = lane & 15, kb = (lane >> 4) * 16;
  int k = kt * 32 + kb + i;
  int o = ct * 16 + col;
  const float* w = (layer == 0) ? w0 : (layer == 1) ? w1 : (layer == 2) ? w2 : w3;
  wt[gid] = (f16)w[o * 256 + k];
}

__global__ __launch_bounds__(256) void gat_main(GatParams p) {
  __shared__ f16   hs[NP * HS];      // node features, row-major [j][o] (f16)
  __shared__ f16   gsT[DD * GST];    // transformed features g, TRANSPOSED [o][j]
  __shared__ f16   at[NP * AS];      // attention matrix [i][j] (f16)
  __shared__ float ssrc[NP];
  __shared__ float sdst[NP];
  __shared__ float xs[NN * 6];

  const int t    = threadIdx.x;
  const int lane = t & 31;
  const int wave = t >> 5;
  const int b    = blockIdx.x;
  const int row  = lane & 15;            // A-fragment / C row group
  const int hi   = lane >> 4;            // 0 = lanes 0-15, 1 = lanes 16-31
  const int kb8  = hi * 8;               // unused (A fragments also use kb16)
  const int kb16 = hi * 16;              // B-fragment K base
  const int col  = lane & 15;            // B/C column

  // ---- stage x into LDS; zero gsT pad columns j=112..127 ----
  for (int idx = t; idx < NN * 6; idx += 256) xs[idx] = p.x[(size_t)b * (NN * 6) + idx];
  for (int idx = t; idx < DD * 16; idx += 256) {
    int o = idx >> 4, c = idx & 15;
    gsT[o * GST + NP + c] = (f16)0.f;
  }
  __syncthreads();

  // ---- remap: hs = relu(x @ rw^T + rb); rows >= NN zeroed ----
  {
    const int o = t;  // blockDim == 256
    float r0 = p.rw[o*6+0], r1 = p.rw[o*6+1], r2 = p.rw[o*6+2];
    float r3 = p.rw[o*6+3], r4 = p.rw[o*6+4], r5 = p.rw[o*6+5];
    float rb = p.rb[o];
    for (int n = 0; n < NP; ++n) {
      float v = 0.f;
      if (n < NN) {
        const float* xr = &xs[n * 6];
        v = rb + xr[0]*r0 + xr[1]*r1 + xr[2]*r2 + xr[3]*r3 + xr[4]*r4 + xr[5]*r5;
        v = v > 0.f ? v : 0.f;
      }
      hs[n * HS + o] = (f16)v;
    }
  }
  __syncthreads();

  const float* asv[4] = {p.as0, p.as1, p.as2, p.as3};
  const float* adv[4] = {p.ad0, p.ad1, p.ad2, p.ad3};
  const float* bbv[4] = {p.bb0, p.bb1, p.bb2, p.bb3};

  for (int layer = 0; layer < 4; ++layer) {
    const f16* wtl = p.wt + layer * 65536;

    // ---- phase 1: g = h @ w^T via WMMA; ct outer (B frags held in regs) ----
    for (int ct = wave; ct < CTN; ct += 8) {
      v16h bw[KT8];
      #pragma unroll
      for (int kt = 0; kt < KT8; ++kt)
        bw[kt] = *(const v16h*)(wtl + (((kt * 16 + ct) * 32 + lane) * 16));
      for (int mt = 0; mt < MT; ++mt) {
        v8f acc = {};
        #pragma unroll
        for (int kt = 0; kt < KT8; ++kt) {
          const f16* ar = &hs[(mt * 16 + row) * HS + kt * 32];
          h8 lo = *(const h8*)(ar + kb16);
          h8 hh = *(const h8*)(ar + kb16 + 8);
          v16h a;
          #pragma unroll
          for (int i = 0; i < 8; ++i) { a[i] = lo[i]; a[i + 8] = hh[i]; }
          acc = __builtin_amdgcn_wmma_f32_16x16x32_f16(false, a, false, bw[kt],
                                                       (short)0, acc, false, false);
        }
        // pack 8 f32 -> 8 f16: consecutive nodes j at fixed feature o -> one b128
        h8 pk;
        #pragma unroll
        for (int r = 0; r < 8; ++r) pk[r] = (f16)acc[r];
        *(h8*)&gsT[(ct * 16 + col) * GST + mt * 16 + hi * 8] = pk;
      }
    }
    __syncthreads();

    // ---- phase 1.5: per-node scores; 2 threads per node, shfl combine ----
    if (t < 2 * NP) {
      const int n = t >> 1, half = t & 1;
      const float* av = asv[layer]; const float* dv = adv[layer];
      float ss = 0.f, sd = 0.f;
      for (int o = half * 128; o < half * 128 + 128; ++o) {
        float gv = (float)gsT[o * GST + n];
        ss += gv * av[o];
        sd += gv * dv[o];
      }
      ss += __shfl_xor(ss, 1);
      sd += __shfl_xor(sd, 1);
      if (half == 0) { ssrc[n] = ss; sdst[n] = sd; }
    }
    __syncthreads();

    // ---- phase 2: softmax rows; 2 threads per row, shfl combine ----
    if (t < 2 * NP) {
      const int i = t >> 1, half = t & 1;
      if (i < NN) {
        const float sd = sdst[i];
        const int j0 = half * 50, j1 = j0 + 50;
        float m = -3.4e38f;
        for (int j = j0; j < j1; ++j) {
          float l = sd + ssrc[j];
          l = l > 0.f ? l : 0.2f * l;
          m = l > m ? l : m;
        }
        float mo = __shfl_xor(m, 1);
        m = mo > m ? mo : m;
        float sum = 0.f;
        for (int j = j0; j < j1; ++j) {
          float l = sd + ssrc[j];
          l = l > 0.f ? l : 0.2f * l;
          float e = __expf(l - m);
          sum += e;
          at[i * AS + j] = (f16)e;
        }
        sum += __shfl_xor(sum, 1);
        float inv = 1.f / sum;
        for (int j = j0; j < j1; ++j)
          at[i * AS + j] = (f16)((float)at[i * AS + j] * inv);
        for (int j = NN + half * 14; j < NN + half * 14 + 14; ++j)
          at[i * AS + j] = (f16)0.f;
      } else {
        for (int j = half * 64; j < half * 64 + 64; ++j)
          at[i * AS + j] = (f16)0.f;
      }
    }
    __syncthreads();

    // ---- phase 3: h' = attn @ g + b (+h, relu); ct outer, B frags in regs ----
    const float* bv = bbv[layer];
    if (layer < 3) {  // prefetch next layer's packed weights (global_prefetch_b8)
      const f16* nxt = p.wt + (layer + 1) * 65536;
      __builtin_prefetch(nxt + t * 256, 0, 1);
    }
    for (int ct = wave; ct < CTN; ct += 8) {
      const int o = ct * 16 + col;
      v16h bg[AKT];
      #pragma unroll
      for (int kt = 0; kt < AKT; ++kt) {
        h8 l0 = *(const h8*)&gsT[o * GST + kt * 32 + kb16];
        h8 l1 = *(const h8*)&gsT[o * GST + kt * 32 + kb16 + 8];
        #pragma unroll
        for (int i = 0; i < 8; ++i) { bg[kt][i] = l0[i]; bg[kt][i + 8] = l1[i]; }
      }
      const float bias = bv[o];
      for (int mt = 0; mt < MT; ++mt) {
        v8f acc = {};
        #pragma unroll
        for (int kt = 0; kt < AKT; ++kt) {
          const f16* ar = &at[(mt * 16 + row) * AS + kt * 32];
          h8 lo = *(const h8*)(ar + kb16);
          h8 hh = *(const h8*)(ar + kb16 + 8);
          v16h a;
          #pragma unroll
          for (int i = 0; i < 8; ++i) { a[i] = lo[i]; a[i + 8] = hh[i]; }
          acc = __builtin_amdgcn_wmma_f32_16x16x32_f16(false, a, false, bg[kt],
                                                       (short)0, acc, false, false);
        }
        #pragma unroll
        for (int r = 0; r < 8; ++r) {
          int node = mt * 16 + r + hi * 8;
          float v = 0.f;
          if (node < NN) {
            v = acc[r] + bias;
            if (layer < 3) {
              v += (float)hs[node * HS + o];   // residual
              v = v > 0.f ? v : 0.f;           // relu
            }
          }
          hs[node * HS + o] = (f16)v;
        }
      }
    }
    __syncthreads();
  }

  // ---- write out[b, 0:100, 0:256] as f32, coalesced ----
  for (int idx = t; idx < NN * DD; idx += 256) {
    int n = idx >> 8, o = idx & 255;
    p.out[(size_t)b * (NN * DD) + idx] = (float)hs[n * HS + o];
  }
}

extern "C" void kernel_launch(void* const* d_in, const int* in_sizes, int n_in,
                              void* d_out, int out_size, void* d_ws, size_t ws_size,
                              hipStream_t stream) {
  (void)in_sizes; (void)n_in; (void)out_size; (void)ws_size;
  GatParams p;
  p.x   = (const float*)d_in[0];
  p.rw  = (const float*)d_in[2];
  p.rb  = (const float*)d_in[3];
  p.wt  = (const f16*)d_ws;
  p.as0 = (const float*)d_in[5];  p.ad0 = (const float*)d_in[6];  p.bb0 = (const float*)d_in[7];
  p.as1 = (const float*)d_in[9];  p.ad1 = (const float*)d_in[10]; p.bb1 = (const float*)d_in[11];
  p.as2 = (const float*)d_in[13]; p.ad2 = (const float*)d_in[14]; p.bb2 = (const float*)d_in[15];
  p.as3 = (const float*)d_in[17]; p.ad3 = (const float*)d_in[18]; p.bb3 = (const float*)d_in[19];
  p.out = (float*)d_out;

  // 4 layers * 128 tiles * 512 halves = 262144 elements
  gat_prepack<<<1024, 256, 0, stream>>>((const float*)d_in[4],  (const float*)d_in[8],
                                        (const float*)d_in[12], (const float*)d_in[16],
                                        (f16*)d_ws);
  gat_main<<<2048, 256, 0, stream>>>(p);
}
